// PEN_21431886807313
// MI455X (gfx1250) — matrix operand order repaired and checked by reading, not compiled, not run on hardware
//
#include <hip/hip_runtime.h>

// ---------------------------------------------------------------------------
// PEN forward for MI455X (gfx1250, wave32, WMMA).
// Model dims (fixed by reference): B=64, D=32, T=64, H=512, L=512, P=5.
// ---------------------------------------------------------------------------

#define B_ 64
#define D_ 32
#define T_ 64
#define H_ 512
#define L_ 512
#define P_ 5

typedef __attribute__((ext_vector_type(16))) _Float16 v16h;
typedef __attribute__((ext_vector_type(8)))  float    v8f;

__device__ __forceinline__ float act_apply(float x, int act) {
  if (act == 1) return tanhf(x);
  if (act == 2) return 1.0f / (1.0f + expf(-x));
  return x;
}

// ---------------------------------------------------------------------------
// Generic WMMA GEMM:  C[M,N] = act( A[M,K] @ W[K,N] + bias0 + bias1 + aux )
//   A row r lives at:  A + (r/aGrpRows)*aGrpStride + (r%aGrpRows)*aRowStride
//   (grpRows=1, strides=0 -> broadcast a single row; grpRows=T -> the
//    [B,D,T,H] text layout for a fixed day)
//   aux (optional) adds aux[(r/auxDiv)*auxLd + col]  (per-batch row term).
//
// Block tile 64x128, 8 waves (wave32), each wave owns a 32x32 patch as 2x2
// v_wmma_f32_16x16x32_f16 accumulators. fp32 -> fp16 conversion happens ONCE
// at LDS staging time; LDS holds fp16 tiles:
//   As[64][40]  row-major (row stride 80B, 16B aligned) -> A fragment = two
//               contiguous 8-half chunks per lane (ds_load_b128).
//   Bt[128][40] B transposed [col][k] -> B fragment = two contiguous 8-half chunks
//               per lane (2x ds_load_b128), no per-wave cvt in the K loop.
// ---------------------------------------------------------------------------
__global__ __launch_bounds__(256) void wmma_gemm_kernel(
    const float* __restrict__ A, int aGrpRows, long aGrpStride, int aRowStride,
    const float* __restrict__ W,
    float* __restrict__ C, int ldc,
    const float* __restrict__ bias0, const float* __restrict__ bias1,
    const float* __restrict__ aux, int auxDiv, int auxLd,
    int M, int N, int K, int act)
{
  __shared__ _Float16 As[64][40];    // 5 KB
  __shared__ _Float16 Bt[128][40];   // 10 KB

  const int tid    = threadIdx.x;
  const int lane   = tid & 31;
  const int wid    = tid >> 5;
  const int wm     = (wid >> 2) * 32;   // wave M origin in tile (0 / 32)
  const int wn     = (wid & 3) * 32;    // wave N origin in tile (0..96)
  const int laneLo = lane & 15;
  const int laneHi = lane >> 4;

  const int mBase = blockIdx.y * 64;
  const int nBase = blockIdx.x * 128;

  v8f acc[2][2] = {};

  // A tile loader: 64 rows x 32 cols, 8 floats/thread
  const int arow = tid >> 2;
  const int acol = (tid & 3) * 8;
  // B tile loader: 32 rows x 128 cols, 16 floats/thread (written transposed)
  const int brow = tid >> 3;
  const int bcol = (tid & 7) * 16;

  // Precompute the A row pointer once (row identity is loop-invariant).
  const int aRow = mBase + arow;
  const float* aptr = nullptr;
  if (aRow < M)
    aptr = A + (long)(aRow / aGrpRows) * aGrpStride
             + (long)(aRow % aGrpRows) * aRowStride;

  for (int k0 = 0; k0 < K; k0 += 32) {
    __syncthreads();   // protect previous iteration's fragment reads

    // ---- stage A tile: fp32 global -> fp16 LDS ----
    if (aptr && (k0 + 32 <= K)) {
      #pragma unroll
      for (int e = 0; e < 8; ++e)
        As[arow][acol + e] = (_Float16)aptr[k0 + acol + e];
    } else {
      #pragma unroll
      for (int e = 0; e < 8; ++e) {
        const int kk = k0 + acol + e;
        As[arow][acol + e] = (aptr && kk < K) ? (_Float16)aptr[kk] : (_Float16)0.0f;
      }
    }
    // ---- stage B tile transposed: fp32 global -> fp16 LDS + prefetch ----
    {
      const int kk = k0 + brow;
      if (kk < K && (nBase + bcol + 16 <= N)) {
        const float* wp = W + (long)kk * N + nBase + bcol;
        #pragma unroll
        for (int e = 0; e < 16; ++e)
          Bt[bcol + e][brow] = (_Float16)wp[e];
      } else {
        #pragma unroll
        for (int e = 0; e < 16; ++e) {
          const int col = nBase + bcol + e;
          Bt[bcol + e][brow] =
              (kk < K && col < N) ? (_Float16)W[(long)kk * N + col] : (_Float16)0.0f;
        }
      }
      if (k0 + 32 < K)   // -> global_prefetch_b8
        __builtin_prefetch(W + (long)(k0 + 32 + brow) * N + nBase + bcol, 0, 1);
    }
    __syncthreads();

    // ---- fragments: contiguous fp16 LDS reads (ISA 7.12.2 layouts) ----
    v16h afr[2], bfr[2];
    #pragma unroll
    for (int i = 0; i < 2; ++i) {
      const _Float16* ap = &As[wm + i * 16 + laneLo][laneHi * 8];
      #pragma unroll
      for (int e = 0; e < 8; ++e) {
        afr[i][e]     = ap[e];        // K = laneHi*8 + e
        afr[i][8 + e] = ap[16 + e];   // K = 16 + laneHi*8 + e
      }
    }
    #pragma unroll
    for (int j = 0; j < 2; ++j) {
      const _Float16* bp = &Bt[wn + j * 16 + laneLo][laneHi * 8];
      #pragma unroll
      for (int e = 0; e < 8; ++e) {
        bfr[j][e]     = bp[e];        // K = laneHi*8 + e
        bfr[j][8 + e] = bp[16 + e];   // K = 16 + laneHi*8 + e
      }
    }
    #pragma unroll
    for (int i = 0; i < 2; ++i)
      #pragma unroll
      for (int j = 0; j < 2; ++j)
        acc[i][j] = __builtin_amdgcn_wmma_f32_16x16x32_f16(
            false, afr[i], false, bfr[j], (short)0, acc[i][j], false, false);
  }

  // ---- epilogue: bias + aux + activation, C/D layout per ISA 7.12.2 ----
  #pragma unroll
  for (int i = 0; i < 2; ++i) {
    #pragma unroll
    for (int j = 0; j < 2; ++j) {
      const int colG = nBase + wn + j * 16 + laneLo;
      #pragma unroll
      for (int v = 0; v < 8; ++v) {
        const int rowG = mBase + wm + i * 16 + v + laneHi * 8;
        if (rowG < M && colG < N) {
          float x = acc[i][j][v];
          if (bias0) x += bias0[colG];
          if (bias1) x += bias1[colG];
          if (aux)   x += aux[(long)(rowG / auxDiv) * auxLd + colG];
          C[(long)rowG * ldc + colG] = act_apply(x, act);
        }
      }
    }
  }
}

// ---------------------------------------------------------------------------
// Wave-level GEMV: out[r*outStride] = act( dot(A[r,:K], w) + b )
// ---------------------------------------------------------------------------
__global__ __launch_bounds__(256) void gemv_kernel(
    const float* __restrict__ A, int lda,
    const float* __restrict__ w, const float* __restrict__ b,
    float* __restrict__ out, int outStride, int M, int K, int act)
{
  const int lane = threadIdx.x & 31;
  const int r = (int)((blockIdx.x * blockDim.x + threadIdx.x) >> 5);
  if (r >= M) return;
  const float* ar = A + (long)r * lda;
  float s = 0.0f;
  for (int k = lane; k < K; k += 32) s += ar[k] * w[k];
  #pragma unroll
  for (int off = 16; off; off >>= 1) s += __shfl_xor(s, off, 32);
  if (lane == 0) out[(long)r * outStride] = act_apply(s + (b ? b[0] : 0.0f), act);
}

// dst[r*dstLd + dstColOff + c] = src-row(r)[c]   (src==nullptr -> zeros)
__global__ void copy_rows_kernel(
    float* __restrict__ dst, int dstLd, int dstColOff,
    const float* __restrict__ src, int grpRows, long grpStride, int rowStride,
    int rows, int cols)
{
  const long i = blockIdx.x * (long)blockDim.x + threadIdx.x;
  if (i >= (long)rows * cols) return;
  const int r = (int)(i / cols), c = (int)(i % cols);
  float v = 0.0f;
  if (src)
    v = src[(long)(r / grpRows) * grpStride + (long)(r % grpRows) * rowStride + c];
  dst[(long)r * dstLd + dstColOff + c] = v;
}

__global__ void zero_kernel(float* __restrict__ p, long n) {
  const long i = blockIdx.x * (long)blockDim.x + threadIdx.x;
  if (i < n) p[i] = 0.0f;
}

// Row softmax, n <= 64, launched <<<rows, 64>>>
__global__ void softmax_kernel(const float* __restrict__ in, long inRowStride,
                               float* __restrict__ out, long outRowStride, int n)
{
  __shared__ float red[64];
  const int r = blockIdx.x, t = threadIdx.x;
  const float v = (t < n) ? in[(long)r * inRowStride + t] : -1e30f;
  red[t] = v; __syncthreads();
  #pragma unroll
  for (int off = 32; off; off >>= 1) {
    if (t < off) red[t] = fmaxf(red[t], red[t + off]);
    __syncthreads();
  }
  const float mx = red[0]; __syncthreads();
  const float e = (t < n) ? expf(v - mx) : 0.0f;
  red[t] = e; __syncthreads();
  #pragma unroll
  for (int off = 32; off; off >>= 1) {
    if (t < off) red[t] += red[t + off];
    __syncthreads();
  }
  if (t < n) out[(long)r * outRowStride + t] = e / red[0];
}

// sel[b,h] = sum_t vos[(b*D+d)*T+t] * text[((b*D+d)*T+t)*H + h]
__global__ void attsum_kernel(const float* __restrict__ vos,
                              const float* __restrict__ text,
                              float* __restrict__ sel, int d)
{
  const int i = blockIdx.x * blockDim.x + threadIdx.x;
  if (i >= B_ * H_) return;
  const int b = i / H_, h = i % H_;
  const long base = ((long)b * D_ + d) * T_;
  float s = 0.0f;
  for (int t = 0; t < T_; ++t)
    s += vos[base + t] * text[(base + t) * (long)H_ + h];
  sel[i] = s;
}

// l = f*l + o*lt  (in place)
__global__ void lcur_kernel(float* __restrict__ l, const float* __restrict__ f,
                            const float* __restrict__ o, const float* __restrict__ lt)
{
  const int i = blockIdx.x * blockDim.x + threadIdx.x;
  if (i >= B_ * H_) return;
  l[i] = f[i] * l[i] + o[i] * lt[i];
}

// h_out[b,h] = d[b]*h_p[b,h] + (1-d[b])*h_l[b,h]
__global__ void hcomb_kernel(float* __restrict__ hout, const float* __restrict__ dg,
                             const float* __restrict__ hp, const float* __restrict__ hl)
{
  const int i = blockIdx.x * blockDim.x + threadIdx.x;
  if (i >= B_ * H_) return;
  const float dd = dg[i / H_];
  hout[i] = dd * hp[i] + (1.0f - dd) * hl[i];
}

// GRU gate combine (PyTorch r,z,n order). gh==nullptr -> use ghvec[3H]
// (decoder: h_prev = 0 so gh reduces to bhh). hprev==nullptr -> zeros.
__global__ void gru_comb_kernel(const float* __restrict__ gi,
                                const float* __restrict__ gh,
                                const float* __restrict__ ghvec,
                                const float* __restrict__ hprev,
                                float* __restrict__ hout)
{
  const int i = blockIdx.x * blockDim.x + threadIdx.x;
  if (i >= B_ * H_) return;
  const int b = i / H_, h = i % H_;
  const long gb = (long)b * 3 * H_;
  const float ghr = gh ? gh[gb + h]          : ghvec[h];
  const float ghz = gh ? gh[gb + H_ + h]     : ghvec[H_ + h];
  const float ghn = gh ? gh[gb + 2 * H_ + h] : ghvec[2 * H_ + h];
  const float r = 1.0f / (1.0f + expf(-(gi[gb + h] + ghr)));
  const float z = 1.0f / (1.0f + expf(-(gi[gb + H_ + h] + ghz)));
  const float n = tanhf(gi[gb + 2 * H_ + h] + r * ghn);
  const float hp = hprev ? hprev[i] : 0.0f;
  hout[i] = (1.0f - z) * n + z * hp;
}

// z[(b*D+d)*L+l] = mu + exp(ls)*eps[(b*D+d)*L+l]
__global__ void z_kernel(const float* __restrict__ mu, const float* __restrict__ ls,
                         const float* __restrict__ eps, float* __restrict__ zout, int d)
{
  const int i = blockIdx.x * blockDim.x + threadIdx.x;
  if (i >= B_ * L_) return;
  const int b = i / L_, l = i % L_;
  const long oi = ((long)b * D_ + d) * L_ + l;
  zout[oi] = mu[i] + expf(ls[i]) * eps[oi];
}

// klpart[d] = -0.5/B * sum_{b,l}(1 + ls - mu^2 - exp(ls))   (single block)
__global__ void kl_part_kernel(const float* __restrict__ mu, const float* __restrict__ ls,
                               float* __restrict__ klpart, int d)
{
  __shared__ float red[256];
  float s = 0.0f;
  for (int i = threadIdx.x; i < B_ * L_; i += 256) {
    const float m = mu[i], v = ls[i];
    s += 1.0f + v - m * m - expf(v);
  }
  red[threadIdx.x] = s; __syncthreads();
  #pragma unroll
  for (int off = 128; off; off >>= 1) {
    if ((int)threadIdx.x < off) red[threadIdx.x] += red[threadIdx.x + off];
    __syncthreads();
  }
  if (threadIdx.x == 0) klpart[d] = (-0.5f / (float)B_) * red[0];
}

__global__ void kl_final_kernel(const float* __restrict__ klpart, float* __restrict__ out) {
  if (threadIdx.x == 0 && blockIdx.x == 0) {
    float s = 0.0f;
    for (int d = 0; d < D_; ++d) s += klpart[d];
    out[0] = s;
  }
}

// sc[b*D+d] = dot(final_pred[b,:], qenc[(d*B+b),:])   (one wave per (b,d))
__global__ __launch_bounds__(256) void tapscore_kernel(
    const float* __restrict__ fp, const float* __restrict__ qenc, float* __restrict__ sc)
{
  const int lane = threadIdx.x & 31;
  const int r = (int)((blockIdx.x * blockDim.x + threadIdx.x) >> 5);
  if (r >= B_ * D_) return;
  const int d = r / B_, b = r % B_;
  const float* x = fp + (long)b * H_;
  const float* y = qenc + ((long)d * B_ + b) * H_;
  float s = 0.0f;
  for (int k = lane; k < H_; k += 32) s += x[k] * y[k];
  #pragma unroll
  for (int off = 16; off; off >>= 1) s += __shfl_xor(s, off, 32);
  if (lane == 0) sc[(long)b * D_ + d] = s;
}

// value[b,h] = sum_d w[b*D+d] * dech[(d*B+b)*H + h]
__global__ void tapvalue_kernel(const float* __restrict__ w, const float* __restrict__ dech,
                                float* __restrict__ val)
{
  const int i = blockIdx.x * blockDim.x + threadIdx.x;
  if (i >= B_ * H_) return;
  const int b = i / H_, h = i % H_;
  float s = 0.0f;
  for (int d = 0; d < D_; ++d)
    s += w[(long)b * D_ + d] * dech[((long)d * B_ + b) * H_ + h];
  val[i] = s;
}

// ---------------------------------------------------------------------------
// Host orchestration
// ---------------------------------------------------------------------------
extern "C" void kernel_launch(void* const* d_in, const int* in_sizes, int n_in,
                              void* d_out, int out_size, void* d_ws, size_t ws_size,
                              hipStream_t stream) {
  (void)in_sizes; (void)n_in; (void)out_size; (void)ws_size;

  // Inputs flattened as a JAX pytree of setup_inputs() (dicts sorted by key):
  // eps, params{dec_gru{Whh,Wih,bhh,bih}, enc_gru{...}, h0, ifu_W6(W,b),
  // ifu_W7, ifu_W8, l0, ls_post, ls_prior, mu_post, mu_prior, predict,
  // tap_Wk, tap_Wq, tap_final, tap_wk, tmu_W3..5, tmu_b3..5, tsu_W1, tsu_W2,
  // tsu_b1, tsu_k}, price_series, text_embeddings
  const float* eps    = (const float*)d_in[0];
  const float* decWhh = (const float*)d_in[1];
  const float* decWih = (const float*)d_in[2];
  const float* decbhh = (const float*)d_in[3];
  const float* decbih = (const float*)d_in[4];
  const float* encWhh = (const float*)d_in[5];
  const float* encWih = (const float*)d_in[6];
  const float* encbhh = (const float*)d_in[7];
  const float* encbih = (const float*)d_in[8];
  const float* h0     = (const float*)d_in[9];
  const float* W6     = (const float*)d_in[10];
  const float* b6     = (const float*)d_in[11];
  const float* W7     = (const float*)d_in[12];
  const float* b7     = (const float*)d_in[13];
  const float* W8     = (const float*)d_in[14];
  const float* b8     = (const float*)d_in[15];
  const float* l0     = (const float*)d_in[16];
  const float* lsW    = (const float*)d_in[17];
  const float* lsb    = (const float*)d_in[18];
  const float* muW    = (const float*)d_in[21];
  const float* mub    = (const float*)d_in[22];
  const float* predW  = (const float*)d_in[25];
  const float* predb  = (const float*)d_in[26];
  const float* tapWq  = (const float*)d_in[29];
  const float* tapbq  = (const float*)d_in[30];
  const float* tapfW  = (const float*)d_in[31];
  const float* tapfb  = (const float*)d_in[32];
  const float* W3     = (const float*)d_in[35];
  const float* b3     = (const float*)d_in[36];
  const float* W4     = (const float*)d_in[37];
  const float* b4     = (const float*)d_in[38];
  const float* W5     = (const float*)d_in[39];
  const float* b5     = (const float*)d_in[40];
  const float* tmub3  = (const float*)d_in[41];
  const float* tmub4  = (const float*)d_in[42];
  const float* tmub5  = (const float*)d_in[43];
  const float* W1     = (const float*)d_in[44];
  const float* b1     = (const float*)d_in[45];
  const float* W2     = (const float*)d_in[46];
  const float* b2     = (const float*)d_in[47];
  const float* tsub1  = (const float*)d_in[48];
  const float* kW     = (const float*)d_in[49];
  const float* kb     = (const float*)d_in[50];
  const float* price  = (const float*)d_in[51];
  const float* text   = (const float*)d_in[52];

  // Output layout (flat, reference return order)
  float* out    = (float*)d_out;
  float* oPred  = out;                                  // [B,D,1]
  float* oZ     = out + (long)B_ * D_;                  // [B,D,L]
  float* oKL    = oZ + (long)B_ * D_ * L_;              // scalar
  float* oVos   = oKL + 1;                              // [B,D,T,1]
  float* oFin   = oVos + (long)B_ * D_ * T_;            // [B,1]

  // Workspace bump allocator
  float* wsp = (float*)d_ws;
  size_t off = 0;
  auto alloc = [&](size_t n) { float* p = wsp + off; off += n; return p; };
  float* scoresT = alloc((size_t)B_ * T_);
  float* attpre  = alloc((size_t)B_ * T_ * H_);
  float* hterm   = alloc((size_t)B_ * H_);
  float* sel     = alloc((size_t)B_ * H_);
  float* ic      = alloc((size_t)B_ * 2 * H_);
  float* fb      = alloc((size_t)B_ * H_);
  float* ob      = alloc((size_t)B_ * H_);
  float* ltb     = alloc((size_t)B_ * H_);
  float* lbuf    = alloc((size_t)B_ * H_);
  float* cin6    = alloc((size_t)B_ * (2 * H_ + P_));
  float* cin7    = alloc((size_t)B_ * 2 * H_);
  float* cin8    = alloc((size_t)B_ * (H_ + P_));
  float* hl      = alloc((size_t)B_ * H_);
  float* hp      = alloc((size_t)B_ * H_);
  float* dg      = alloc((size_t)B_);
  float* hser    = alloc((size_t)D_ * B_ * H_);
  float* ench    = alloc((size_t)B_ * H_);
  float* gi      = alloc((size_t)B_ * 3 * H_);
  float* gh      = alloc((size_t)B_ * 3 * H_);
  float* ei      = alloc((size_t)B_ * (H_ + L_));
  float* mu      = alloc((size_t)B_ * L_);
  float* ls      = alloc((size_t)B_ * L_);
  float* xdec    = alloc((size_t)B_ * (H_ + L_));
  float* dech    = alloc((size_t)D_ * B_ * H_);
  float* klp     = alloc((size_t)D_);
  float* qenc    = alloc((size_t)D_ * B_ * H_);
  float* tsc     = alloc((size_t)B_ * D_);
  float* tw      = alloc((size_t)B_ * D_);
  float* val     = alloc((size_t)B_ * H_);
  float* cfin    = alloc((size_t)B_ * 2 * H_);

  auto gemm = [&](const float* A, int aGrp, long aGs, int aRs, const float* Wm,
                  float* C, int ldc, const float* bb0, const float* bb1,
                  const float* aux, int auxDiv, int auxLd,
                  int M, int N, int K, int act) {
    dim3 g((N + 127) / 128, (M + 63) / 64);
    wmma_gemm_kernel<<<g, 256, 0, stream>>>(A, aGrp, aGs, aRs, Wm, C, ldc, bb0, bb1,
                                            aux, auxDiv ? auxDiv : 1, auxLd, M, N, K, act);
  };
  auto gemv = [&](const float* A, int lda, const float* wv, const float* bv,
                  float* o, int ostr, int M, int K, int act) {
    gemv_kernel<<<(M + 7) / 8, 256, 0, stream>>>(A, lda, wv, bv, o, ostr, M, K, act);
  };
  auto copyr = [&](float* dst, int dLd, int dOff, const float* src, int grp,
                   long gs, int rs, int rows, int cols) {
    long n = (long)rows * cols;
    copy_rows_kernel<<<(int)((n + 255) / 256), 256, 0, stream>>>(dst, dLd, dOff, src,
                                                                 grp, gs, rs, rows, cols);
  };
  const int GBH = (B_ * H_ + 255) / 256;

  // ---- init carries ----
  copyr(lbuf, H_, 0, l0, 1, 0, 0, B_, H_);                          // l0 broadcast
  zero_kernel<<<GBH, 256, 0, stream>>>(ench, (long)B_ * H_);

  // =======================  SRL + DRG fused day loop  =======================
  for (int d = 0; d < D_; ++d) {
    const float* hprev = d ? hser + (size_t)(d - 1) * B_ * H_ : h0;
    const int hpGrp = d ? B_ : 1;          // grpRows=1 => broadcast h0 row
    const int hpRs  = d ? H_ : 0;

    // --- TSU attention ---
    gemm(hprev, hpGrp, 0, hpRs, W1, hterm, H_, b1, nullptr,
         nullptr, 1, 0, B_, H_, H_, 0);                              // h@W1+b1
    gemm(text + (size_t)d * T_ * H_, T_, (long)D_ * T_ * H_, H_, W2,
         attpre, H_, b2, tsub1, hterm, T_, H_,
         B_ * T_, H_, H_, 1);                                        // tanh(text@W2+..+hterm)
    gemv(attpre, H_, kW, kb, scoresT, 1, B_ * T_, H_, 0);
    softmax_kernel<<<B_, 64, 0, stream>>>(scoresT, T_, oVos + (long)d * T_,
                                          (long)D_ * T_, T_);        // vos -> d_out
    attsum_kernel<<<GBH, 256, 0, stream>>>(oVos, text, sel, d);

    // --- TMU gates ---
    copyr(ic, 2 * H_, 0, sel, B_, 0, H_, B_, H_);
    copyr(ic, 2 * H_, H_, hprev, hpGrp, 0, hpRs, B_, H_);
    gemm(ic, B_, 0, 2 * H_, W3, fb,  H_, b3, tmub3, nullptr, 1, 0, B_, H_, 2 * H_, 2);
    gemm(ic, B_, 0, 2 * H_, W4, ob,  H_, b4, tmub4, nullptr, 1, 0, B_, H_, 2 * H_, 2);
    gemm(ic, B_, 0, 2 * H_, W5, ltb, H_, b5, tmub5, nullptr, 1, 0, B_, H_, 2 * H_, 1);
    lcur_kernel<<<GBH, 256, 0, stream>>>(lbuf, fb, ob, ltb);

    // --- IFU ---
    copyr(cin6, 2 * H_ + P_, 0, price + (size_t)d * P_, 1, (long)D_ * P_, 0, B_, P_);
    copyr(cin6, 2 * H_ + P_, P_, lbuf, B_, 0, H_, B_, H_);
    copyr(cin6, 2 * H_ + P_, P_ + H_, hprev, hpGrp, 0, hpRs, B_, H_);
    gemv(cin6, 2 * H_ + P_, W6, b6, dg, 1, B_, 2 * H_ + P_, 2);
    copyr(cin7, 2 * H_, 0, lbuf, B_, 0, H_, B_, H_);
    copyr(cin7, 2 * H_, H_, hprev, hpGrp, 0, hpRs, B_, H_);
    gemm(cin7, B_, 0, 2 * H_, W7, hl, H_, b7, nullptr, nullptr, 1, 0, B_, H_, 2 * H_, 1);
    copyr(cin8, H_ + P_, 0, price + (size_t)d * P_, 1, (long)D_ * P_, 0, B_, P_);
    copyr(cin8, H_ + P_, P_, hprev, hpGrp, 0, hpRs, B_, H_);
    gemm(cin8, B_, 0, H_ + P_, W8, hp, H_, b8, nullptr, nullptr, 1, 0, B_, H_, H_ + P_, 1);
    hcomb_kernel<<<GBH, 256, 0, stream>>>(hser + (size_t)d * B_ * H_, dg, hp, hl);

    // --- DRG: encoder GRU ---
    gemm(hser + (size_t)d * B_ * H_, B_, 0, H_, encWih, gi, 3 * H_,
         encbih, nullptr, nullptr, 1, 0, B_, 3 * H_, H_, 0);
    gemm(ench, B_, 0, H_, encWhh, gh, 3 * H_,
         encbhh, nullptr, nullptr, 1, 0, B_, 3 * H_, H_, 0);
    gru_comb_kernel<<<GBH, 256, 0, stream>>>(gi, gh, nullptr, ench, ench);

    // --- posterior + reparameterize + KL ---
    copyr(ei, H_ + L_, 0, ench, B_, 0, H_, B_, H_);
    copyr(ei, H_ + L_, H_, d ? oZ + (size_t)(d - 1) * L_ : nullptr,
          1, (long)D_ * L_, 0, B_, L_);                              // z_prev (0 at d=0)
    gemm(ei, B_, 0, H_ + L_, muW, mu, L_, mub, nullptr, nullptr, 1, 0, B_, L_, H_ + L_, 0);
    gemm(ei, B_, 0, H_ + L_, lsW, ls, L_, lsb, nullptr, nullptr, 1, 0, B_, L_, H_ + L_, 0);
    z_kernel<<<(B_ * L_ + 255) / 256, 256, 0, stream>>>(mu, ls, eps, oZ, d);
    kl_part_kernel<<<1, 256, 0, stream>>>(mu, ls, klp, d);

    // --- decoder GRU (hidden starts at zero each step) + prediction ---
    copyr(xdec, H_ + L_, 0, hser + (size_t)d * B_ * H_, B_, 0, H_, B_, H_);
    copyr(xdec, H_ + L_, H_, oZ + (size_t)d * L_, 1, (long)D_ * L_, 0, B_, L_);
    gemm(xdec, B_, 0, H_ + L_, decWih, gi, 3 * H_,
         decbih, nullptr, nullptr, 1, 0, B_, 3 * H_, H_ + L_, 0);
    gru_comb_kernel<<<GBH, 256, 0, stream>>>(gi, nullptr, decbhh, nullptr,
                                             dech + (size_t)d * B_ * H_);
    gemv(dech + (size_t)d * B_ * H_, H_, predW, predb, oPred + d, D_, B_, H_, 2);
  }

  kl_final_kernel<<<1, 32, 0, stream>>>(klp, oKL);

  // =======================  Temporal attention epilogue  ====================
  const float* finalPred = dech + (size_t)(D_ - 1) * B_ * H_;
  gemm(dech, D_ * B_, 0, H_, tapWq, qenc, H_, tapbq, nullptr,
       nullptr, 1, 0, D_ * B_, H_, H_, 1);                           // tanh(dech@Wq+bq)
  tapscore_kernel<<<(B_ * D_) / 8, 256, 0, stream>>>(finalPred, qenc, tsc);
  softmax_kernel<<<B_, 64, 0, stream>>>(tsc, D_, tw, D_, D_);
  tapvalue_kernel<<<GBH, 256, 0, stream>>>(tw, dech, val);
  copyr(cfin, 2 * H_, 0, val, B_, 0, H_, B_, H_);
  copyr(cfin, 2 * H_, H_, finalPred, B_, 0, H_, B_, H_);
  gemv(cfin, 2 * H_, tapfW, tapfb, oFin, 1, B_, 2 * H_, 2);
}
